// KPConv_24790551232572
// MI455X (gfx1250) — matrix operand order repaired and checked by reading, not compiled, not run on hardware
//
#include <hip/hip_runtime.h>

typedef __attribute__((ext_vector_type(2)))  _Float16 v2h;
typedef __attribute__((ext_vector_type(8)))  _Float16 v8h;
typedef __attribute__((ext_vector_type(16))) _Float16 v16h;
typedef __attribute__((ext_vector_type(8)))  float    v8f;

#define H_NB   40
#define KPTS   15
#define CIN    64
#define COUT   128
#define KDIM   (KPTS * CIN)    // 960 : collapsed reduction dim
#define KSTEPS (KDIM / 32)     // 30  : WMMA K=32 steps
#define ROWP   968             // padded LDS row stride (halves) -> conflict-free b128 A loads
#define KP_EXTENT_INV (1.0f / 1.2f)

// ---- Prepass: per-support-row positivity flag (for neighbor_num heuristic) ----
__global__ void kpconv_prep_flags(const float* __restrict__ x,
                                  float* __restrict__ flags, int M) {
    int m = blockIdx.x * blockDim.x + threadIdx.x;
    if (m >= M) return;
    const float4* xr = (const float4*)(x + (size_t)m * CIN);
    float s = 0.f;
#pragma unroll
    for (int i = 0; i < CIN / 4; ++i) { float4 v = xr[i]; s += v.x + v.y + v.z + v.w; }
    flags[m] = (s > 0.f) ? 1.f : 0.f;
}

// ---- Prepass: pack weights f32 -> f16 in WMMA B-fragment order ----
// B fragment (16x16x32 f16): lane reads 16 contiguous f16, element j <-> K = s*32 + (j<8 ? (lane>>4)*8 + j : 16 + (lane>>4)*8 + j-8)
__global__ void kpconv_prep_bpack(const float* __restrict__ w,
                                  _Float16* __restrict__ bp) {
    int e = blockIdx.x * blockDim.x + threadIdx.x;
    if (e >= KDIM * COUT) return;
    int j    = e & 15;
    int lane = (e >> 4) & 31;
    int rest = e >> 9;
    int s    = rest % KSTEPS;
    int t    = rest / KSTEPS;
    int kd   = s * 32 + ((j < 8) ? ((lane >> 4) * 8 + j) : (16 + (lane >> 4) * 8 + (j - 8)));       // collapsed (k*64 + c)
    int co   = t * 16 + (lane & 15);
    bp[e] = (_Float16)w[(size_t)kd * COUT + co];      // weights flat: ((k*64+c)*128 + o)
}

// ---- Main: 16 query points per block, 8 waves; phase1 gather -> LDS f16, phase2 WMMA ----
__global__ __launch_bounds__(256) void kpconv_main(
    const float* __restrict__ q_pts, const float* __restrict__ s_pts,
    const int* __restrict__ inds,    const float* __restrict__ x,
    const float* __restrict__ kp,    const _Float16* __restrict__ bpack,
    const float* __restrict__ flags, float* __restrict__ out, int N, int M) {

    __shared__ __attribute__((aligned(32))) _Float16 wfh[16 * ROWP];
    __shared__ __attribute__((aligned(16))) float inv_cnt[16];

    const int lane = threadIdx.x & 31;
    const int wave = threadIdx.x >> 5;
    const int n0   = blockIdx.x * 16;

    // one kernel point per lane (lanes 0..14)
    float kx = 0.f, ky = 0.f, kz = 0.f, kq = 0.f;
    if (lane < KPTS) {
        kx = kp[lane * 3 + 0]; ky = kp[lane * 3 + 1]; kz = kp[lane * 3 + 2];
        kq = kx * kx + ky * ky + kz * kz;
    }

    // ---------------- Phase 1: weighted feature gather (2 points per wave) ----------------
    for (int pp = 0; pp < 2; ++pp) {
        const int p = wave * 2 + pp;
        const int n = n0 + p;
        float acc[KPTS][2];
#pragma unroll
        for (int k = 0; k < KPTS; ++k) { acc[k][0] = 0.f; acc[k][1] = 0.f; }
        float cnt = 0.f;

        if (n < N) {
            const float qx = q_pts[n * 3 + 0];
            const float qy = q_pts[n * 3 + 1];
            const float qz = q_pts[n * 3 + 2];
            for (int h = 0; h < H_NB; ++h) {
                const int idx = inds[n * H_NB + h];         // uniform across wave
                if (idx < M) {                              // idx==M -> shadow: w=0, feat=0
                    const float sx = s_pts[idx * 3 + 0] - qx;
                    const float sy = s_pts[idx * 3 + 1] - qy;
                    const float sz = s_pts[idx * 3 + 2] - qz;
                    const float n2 = sx * sx + sy * sy + sz * sz;
                    float wk = 0.f;
                    if (lane < KPTS) {
                        float sq = n2 + kq - 2.f * (sx * kx + sy * ky + sz * kz);
                        sq = fmaxf(sq, 0.f);
                        wk = fmaxf(1.f - sqrtf(sq) * KP_EXTENT_INV, 0.f);
                    }
                    const float2 f = ((const float2*)(x + (size_t)idx * CIN))[lane];
                    cnt += flags[idx];                       // same value in every lane
#pragma unroll
                    for (int k = 0; k < KPTS; ++k) {
                        const float wb = __shfl(wk, k, 32);
                        acc[k][0] += wb * f.x;
                        acc[k][1] += wb * f.y;
                    }
                }
            }
        }
#pragma unroll
        for (int k = 0; k < KPTS; ++k) {
            v2h pr = { (_Float16)acc[k][0], (_Float16)acc[k][1] };
            *(v2h*)&wfh[p * ROWP + k * CIN + 2 * lane] = pr;
        }
        if (lane == 0) inv_cnt[p] = 1.f / fmaxf(cnt, 1.f);
    }
    __syncthreads();

    // ---------------- Phase 2: [16 x 960] x [960 x 16] per wave via WMMA ----------------
    const int t    = wave;           // cout tile
    const int half = lane >> 4;
    const int m    = lane & 15;
    v8f c = {0.f, 0.f, 0.f, 0.f, 0.f, 0.f, 0.f, 0.f};
    const int aRow = m * ROWP;
    const int h8   = half * 8;
#pragma unroll 2
    for (int s = 0; s < KSTEPS; ++s) {
        // A fragment (16-bit A 16x32): elems 0..7 = K s*32+h8.., elems 8..15 = K s*32+16+h8..
        const v8h alo = *(const v8h*)&wfh[aRow + s * 32 + h8];
        const v8h ahi = *(const v8h*)&wfh[aRow + s * 32 + 16 + h8];
        const v16h a = __builtin_shufflevector(alo, ahi,
            0, 1, 2, 3, 4, 5, 6, 7, 8, 9, 10, 11, 12, 13, 14, 15);
        const v16h b = *(const v16h*)(bpack + (((size_t)t * KSTEPS + s) * 32 + lane) * 16);
        c = __builtin_amdgcn_wmma_f32_16x16x32_f16(false, a, false, b, (short)0, c,
                                                   false, false);
    }

    // ---------------- Epilogue: normalize + store (C: VGPR r -> M = r + 8*half) ----------
    const float4 ic0 = *(const float4*)&inv_cnt[8 * half];       // one ds_load_b128
    const float4 ic1 = *(const float4*)&inv_cnt[8 * half + 4];   // one ds_load_b128
    const float icv[8] = {ic0.x, ic0.y, ic0.z, ic0.w, ic1.x, ic1.y, ic1.z, ic1.w};
    float* op = out + (size_t)(n0 + 8 * half) * COUT + t * 16 + m;
    if (n0 + 16 <= N) {                       // block-uniform fast path: no EXEC churn
#pragma unroll
        for (int r = 0; r < 8; ++r) op[(size_t)r * COUT] = c[r] * icv[r];
    } else {                                  // ragged tail fallback
#pragma unroll
        for (int r = 0; r < 8; ++r)
            if (n0 + 8 * half + r < N) op[(size_t)r * COUT] = c[r] * icv[r];
    }
}

extern "C" void kernel_launch(void* const* d_in, const int* in_sizes, int n_in,
                              void* d_out, int out_size, void* d_ws, size_t ws_size,
                              hipStream_t stream) {
    const float* q    = (const float*)d_in[0];
    const float* s    = (const float*)d_in[1];
    const int*   inds = (const int*)d_in[2];
    const float* x    = (const float*)d_in[3];
    const float* kp   = (const float*)d_in[4];
    const float* w    = (const float*)d_in[5];
    float* out = (float*)d_out;
    const int N = in_sizes[0] / 3;
    const int M = in_sizes[1] / 3;

    // workspace: [ Bpack f16 (245760 B) | pad to 256 | flags f32 (4*M B) ]
    _Float16* bpack = (_Float16*)d_ws;
    size_t off = ((size_t)KDIM * COUT * sizeof(_Float16) + 255) & ~(size_t)255;
    float* flags = (float*)((char*)d_ws + off);

    kpconv_prep_bpack<<<(KDIM * COUT + 255) / 256, 256, 0, stream>>>(w, bpack);
    kpconv_prep_flags<<<(M + 255) / 256, 256, 0, stream>>>(x, flags, M);
    kpconv_main<<<(N + 15) / 16, 256, 0, stream>>>(q, s, inds, x, kp, bpack, flags,
                                                   out, N, M);
}
